// GNNLF_87445534146863
// MI455X (gfx1250) — matrix operand order repaired and checked by reading, not compiled, not run on hardware
//
#include <hip/hip_runtime.h>
#include <stdint.h>

#define BB   16
#define NN   128
#define HH   128
#define EFD  64
#define NLAYER 4

typedef __attribute__((ext_vector_type(16))) __bf16         v16bf;
typedef __attribute__((ext_vector_type(8)))  float          f32x8;
typedef __attribute__((ext_vector_type(4)))  float          f32x4;
typedef __attribute__((ext_vector_type(4)))  unsigned int   u32x4;

__device__ __forceinline__ unsigned short f2bf(float f) {
    unsigned int u = __float_as_uint(f);
    u += 0x7fffu + ((u >> 16) & 1u);          // round-to-nearest-even
    return (unsigned short)(u >> 16);
}
__device__ __forceinline__ float bf2f(unsigned short h) {
    return __uint_as_float(((unsigned int)h) << 16);
}
__device__ __forceinline__ float silu_f(float x) {
    return x * __builtin_amdgcn_rcpf(1.0f + __expf(-x));
}

// ---- fragment packing ------------------------------------------------------
// LDS operands are "fragment-packed": per 32-wide K step, a 512-u16 block laid
// out [lane 0..31][elem 0..15]; a lane fetches its fragment with 2 ds_load_b128.
__device__ __forceinline__ int a_pack_idx(int m, int kp) {      // A (16x32 MxK)
    const int g = kp >> 4, kg = kp & 15, half = kg >> 3;
    const int r = g * 4 + ((kg & 7) >> 1);
    return (m + 16 * half) * 16 + 2 * r + (kp & 1);
}
__device__ __forceinline__ int b_pack_idx(int kp, int n) {      // B (32x16 KxN)
    const int half = (kp >> 3) & 1, r = (kp >> 4) * 4 + ((kp & 7) >> 1);
    return (n + 16 * half) * 16 + 2 * r + (kp & 1);
}
__device__ __forceinline__ v16bf load_frag_packed(const unsigned short* p, int lane) {
    const u32x4* q = (const u32x4*)(p + lane * 16);
    union { u32x4 v[2]; v16bf f; } u;
    u.v[0] = q[0]; u.v[1] = q[1];
    return u.f;
}
__device__ __forceinline__ f32x8 wmma_bf16(v16bf a, v16bf b, f32x8 c) {
    return __builtin_amdgcn_wmma_f32_16x16x32_bf16(false, a, false, b, (short)0, c, false, false);
}

// ---- fp32 -> bf16 weight conversion into fragment-packed layout ------------
// dst block order: (nt * Ksteps + kt) * 512
__global__ void k_cvt_pack(const float* __restrict__ src, unsigned short* __restrict__ dst,
                           int K, int N) {
    int i = blockIdx.x * 256 + threadIdx.x;
    if (i >= K * N) return;
    const int k = i / N, n = i - k * N;
    const int Ks = K >> 5;
    dst[((size_t)((n >> 4) * Ks + (k >> 5))) * 512 + b_pack_idx(k & 31, n & 15)] = f2bf(src[i]);
}

// ---- sn = LayerNorm(emb[z]) ------------------------------------------------
__global__ void k_sn(const int* __restrict__ z, const float* __restrict__ emb, float* __restrict__ snW) {
    const int bn = blockIdx.x, t = threadIdx.x;
    const int zi = z[bn];
    float x = emb[(size_t)zi * HH + t];
    __shared__ float red[HH];
    red[t] = x; __syncthreads();
    for (int s = 64; s > 0; s >>= 1) { if (t < s) red[t] += red[t + s]; __syncthreads(); }
    float mean = red[0] / HH; __syncthreads();
    red[t] = x * x; __syncthreads();
    for (int s = 64; s > 0; s >>= 1) { if (t < s) red[t] += red[t + s]; __syncthreads(); }
    float var = red[0] / HH - mean * mean;
    snW[(size_t)bn * HH + t] = (x - mean) * __builtin_amdgcn_rsqf(var + 1e-5f);
}

// ---- mask = silu(ef@Wef+bef)*ea (bf16) and s += sum_j mask*sn --------------
__global__ void k_mask_nbr(const float* __restrict__ ef, const float* __restrict__ ea,
                           const float* __restrict__ s_in, const float* __restrict__ snW,
                           const unsigned short* __restrict__ WefB, const float* __restrict__ bef,
                           unsigned short* __restrict__ maskW, float* __restrict__ sW) {
    __shared__ __align__(16) unsigned short sWef[EFD * HH];   // packed (nt*2+kt)*512
    __shared__ __align__(16) unsigned short sA[2 * 512];      // packed A, 2 k-steps
    __shared__ __align__(16) float sSn[16 * HH];
    const int tid = threadIdx.x, lane = tid & 31, wid = tid >> 5;
    const int bi = blockIdx.x, b = bi / NN;
    const size_t rowB = (size_t)bi * NN;
    for (int t = tid; t < (EFD * HH) / 8; t += 256)
        ((u32x4*)sWef)[t] = ((const u32x4*)WefB)[t];
    float accS = 0.0f;
    for (int jt = 0; jt < 8; ++jt) {
        __syncthreads();
        {   // A tile: 16 rows x 64 K, contiguous in ef; scatter to packed
            const float* efb = ef + (rowB + jt * 16) * EFD;
            for (int t = tid; t < 16 * EFD; t += 256) {
                const int r = t >> 6, k = t & 63;
                sA[(k >> 5) * 512 + a_pack_idx(r, k & 31)] = f2bf(efb[t]);
            }
        }
        {   // sn tile: contiguous 16*128 floats
            const f32x4* src = (const f32x4*)(snW + ((size_t)b * NN + jt * 16) * HH);
            for (int t = tid; t < (16 * HH) / 4; t += 256) ((f32x4*)sSn)[t] = src[t];
        }
        if (jt + 1 < 8 && tid < 16)
            __builtin_prefetch(ef + (rowB + (jt + 1) * 16 + tid) * EFD, 0, 1);
        __syncthreads();
        // preload all fragments (2 k-steps), then back-to-back WMMA
        v16bf a0 = load_frag_packed(sA, lane);
        v16bf b0 = load_frag_packed(sWef + (wid * 2) * 512, lane);
        v16bf a1 = load_frag_packed(sA + 512, lane);
        v16bf b1 = load_frag_packed(sWef + (wid * 2 + 1) * 512, lane);
        f32x8 acc = {};
        acc = wmma_bf16(a0, b0, acc);
        acc = wmma_bf16(a1, b1, acc);
        const int half = lane >> 4, col = wid * 16 + (lane & 15);
        const float bv = bef[col];
#pragma unroll
        for (int r = 0; r < 8; ++r) {
            const int row = r + 8 * half, j = jt * 16 + row;
            float m = silu_f(acc[r] + bv) * ea[rowB + j];
            maskW[(rowB + j) * HH + col] = f2bf(m);
            accS += m * sSn[row * HH + col];
        }
    }
    accS += __shfl_xor(accS, 16);
    if (lane < 16) {
        const int h = wid * 16 + lane;
        sW[(size_t)bi * HH + h] = s_in[(size_t)bi * HH + h] + accS;
    }
}

// ---- generic node GEMM: Y = epi(X@W + b), X (M,128), W (128,128) bf16 ------
// mode 0: none  1: silu(LN(.))  2: silu(.)
__global__ void k_node_gemm(const float* __restrict__ X, const unsigned short* __restrict__ Wb,
                            const float* __restrict__ bias, float* __restrict__ Y, int mode) {
    __shared__ __align__(16) unsigned short sWgt[HH * HH];    // packed (nt*4+kt)*512
    __shared__ __align__(16) unsigned short sA[4 * 512];
    __shared__ float sC[16 * HH];
    __shared__ float mu[16], iv[16];
    const int tid = threadIdx.x, lane = tid & 31, wid = tid >> 5;
    const int m0 = blockIdx.x * 16;
    for (int t = tid; t < (HH * HH) / 8; t += 256)
        ((u32x4*)sWgt)[t] = ((const u32x4*)Wb)[t];
    {
        const float* xb = X + (size_t)m0 * HH;
        for (int t = tid; t < 16 * HH; t += 256) {
            const int r = t >> 7, k = t & 127;
            sA[(k >> 5) * 512 + a_pack_idx(r, k & 31)] = f2bf(xb[t]);
        }
    }
    __syncthreads();
    // double-buffered fragment pipeline over 4 k-steps
    f32x8 acc = {};
    v16bf a = load_frag_packed(sA, lane);
    v16bf bb = load_frag_packed(sWgt + (wid * 4) * 512, lane);
#pragma unroll
    for (int kt = 0; kt < 4; ++kt) {
        v16bf an, bn;
        if (kt < 3) {
            an = load_frag_packed(sA + (kt + 1) * 512, lane);
            bn = load_frag_packed(sWgt + (wid * 4 + kt + 1) * 512, lane);
        }
        acc = wmma_bf16(a, bb, acc);
        if (kt < 3) { a = an; bb = bn; }
    }
    const int half = lane >> 4, col = wid * 16 + (lane & 15);
    const float bv = bias ? bias[col] : 0.0f;
    if (mode == 1) {
#pragma unroll
        for (int r = 0; r < 8; ++r) { int row = r + 8 * half; sC[row * HH + col] = acc[r] + bv; }
        __syncthreads();
        if (tid < 16) {
            float s = 0.f, q = 0.f;
            for (int c = 0; c < HH; ++c) { float v = sC[tid * HH + c]; s += v; q += v * v; }
            float m = s / HH; mu[tid] = m;
            iv[tid] = __builtin_amdgcn_rsqf(q / HH - m * m + 1e-5f);
        }
        __syncthreads();
#pragma unroll
        for (int r = 0; r < 8; ++r) {
            int row = r + 8 * half;
            Y[(size_t)(m0 + row) * HH + col] = silu_f((sC[row * HH + col] - mu[row]) * iv[row]);
        }
    } else {
#pragma unroll
        for (int r = 0; r < 8; ++r) {
            int row = r + 8 * half;
            float v = acc[r] + bv;
            Y[(size_t)(m0 + row) * HH + col] = (mode == 2) ? silu_f(v) : v;
        }
    }
}

// ---- CFConvS2V: v[bi,d,h] = sum_j s1[b,j,h]*mask[bi,j,h]*ev[bi,j,d] --------
__global__ void k_v(const unsigned short* __restrict__ maskW, const float* __restrict__ s1W,
                    const float* __restrict__ ev, float* __restrict__ vW) {
    const int bi = blockIdx.x, b = bi / NN;
    const int tid = threadIdx.x, hp = tid & 63, qd = tid >> 6;
    const int h0 = hp * 2;
    const size_t rowB = (size_t)bi * NN;
    float acc[6] = {0.f, 0.f, 0.f, 0.f, 0.f, 0.f};
    for (int j = qd * 32; j < qd * 32 + 32; ++j) {
        const unsigned int mm = *(const unsigned int*)(maskW + (rowB + j) * HH + h0);
        const float* sp = s1W + ((size_t)b * NN + j) * HH + h0;
        const float w0 = bf2f((unsigned short)mm) * sp[0];
        const float w1 = bf2f((unsigned short)(mm >> 16)) * sp[1];
        const float* e = ev + (rowB + j) * 3;
        acc[0] += w0 * e[0]; acc[1] += w0 * e[1]; acc[2] += w0 * e[2];
        acc[3] += w1 * e[0]; acc[4] += w1 * e[1]; acc[5] += w1 * e[2];
    }
    __shared__ float red[4][3][HH];
#pragma unroll
    for (int d = 0; d < 3; ++d) { red[qd][d][h0] = acc[d]; red[qd][d][h0 + 1] = acc[3 + d]; }
    __syncthreads();
    if (qd == 0) {
#pragma unroll
        for (int d = 0; d < 3; ++d) {
            float t0 = red[0][d][h0] + red[1][d][h0] + red[2][d][h0] + red[3][d][h0];
            float t1 = red[0][d][h0 + 1] + red[1][d][h0 + 1] + red[2][d][h0 + 1] + red[3][d][h0 + 1];
            vW[(size_t)bi * 384 + d * HH + h0]     = t0;
            vW[(size_t)bi * 384 + d * HH + h0 + 1] = t1;
        }
    }
}

// ---- edge MLP: dire = silu(silu([dir2|dir3]*ea @Wd1+bd1)@Wd2+bd2); mask *= dire
__global__ void k_edge(const float* __restrict__ ea, const float* __restrict__ ev,
                       const float* __restrict__ qW, const float* __restrict__ kW,
                       const float* __restrict__ vW,
                       const unsigned short* __restrict__ Wd1B, const float* __restrict__ bd1,
                       const unsigned short* __restrict__ Wd2B, const float* __restrict__ bd2,
                       unsigned short* __restrict__ maskW) {
    extern __shared__ unsigned short smem[];
    unsigned short* Wd1s = smem;                  // packed (nt*8+kt)*512, 128KB
    unsigned short* Wd2s = Wd1s + 256 * 256;      // packed (nt*8+kt)*512, 64KB
    unsigned short* At   = Wd2s + 256 * 128;      // packed A, 8 k-steps
    unsigned short* H1t  = At + 8 * 512;          // packed A for GEMM2, 8 k-steps
    float* vk = (float*)(H1t + 8 * 512);          // v[i] (384) + k[i] (384)
    const int tid = threadIdx.x, lane = tid & 31, wid = tid >> 5;
    const int bi = blockIdx.x, b = bi / NN;
    const size_t rowB = (size_t)bi * NN;
    for (int t = tid; t < (256 * 256) / 8; t += 256) ((u32x4*)Wd1s)[t] = ((const u32x4*)Wd1B)[t];
    for (int t = tid; t < (256 * 128) / 8; t += 256) ((u32x4*)Wd2s)[t] = ((const u32x4*)Wd2B)[t];
    for (int t = tid; t < 384; t += 256) {
        vk[t]       = vW[(size_t)bi * 384 + t];
        vk[384 + t] = kW[(size_t)bi * 384 + t];
    }
    __syncthreads();
    const int half = lane >> 4, n15 = lane & 15;
    for (int jt = 0; jt < 8; ++jt) {
        // A tile (16 x 256): [dir2 | dir3] * ea -> packed bf16
        for (int t = tid; t < 16 * 256; t += 256) {
            const int jr = t >> 8, c = t & 255, h = c & 127;
            const int j = jt * 16 + jr;
            const float e = ea[rowB + j];
            float val;
            if (c < 128) {
                const float* ed = ev + (rowB + j) * 3;
                val = vk[h] * ed[0] + vk[128 + h] * ed[1] + vk[256 + h] * ed[2];
            } else {
                const float* qj = qW + ((size_t)b * NN + j) * 384 + h;
                val = qj[0] * vk[384 + h] + qj[128] * vk[512 + h] + qj[256] * vk[640 + h];
            }
            At[(c >> 5) * 512 + a_pack_idx(jr, c & 31)] = f2bf(val * e);
        }
        if (jt + 1 < 8 && tid < 16)
            __builtin_prefetch(qW + ((size_t)b * NN + (jt + 1) * 16 + tid) * 384, 0, 1);
        __syncthreads();
        // GEMM1: (16x256)@(256x256) -> H1 (silu), one A-frag feeds two N-subtiles;
        // double-buffered so LDS latency overlaps the WMMAs.
        {
            const int nt0 = wid * 2, nt1 = nt0 + 1;
            f32x8 acc0 = {}, acc1 = {};
            v16bf a  = load_frag_packed(At, lane);
            v16bf b0 = load_frag_packed(Wd1s + (nt0 * 8) * 512, lane);
            v16bf b1 = load_frag_packed(Wd1s + (nt1 * 8) * 512, lane);
#pragma unroll
            for (int kt = 0; kt < 8; ++kt) {
                v16bf an, b0n, b1n;
                if (kt < 7) {
                    an  = load_frag_packed(At + (kt + 1) * 512, lane);
                    b0n = load_frag_packed(Wd1s + (nt0 * 8 + kt + 1) * 512, lane);
                    b1n = load_frag_packed(Wd1s + (nt1 * 8 + kt + 1) * 512, lane);
                }
                acc0 = wmma_bf16(a, b0, acc0);
                acc1 = wmma_bf16(a, b1, acc1);
                if (kt < 7) { a = an; b0 = b0n; b1 = b1n; }
            }
            const int col0 = nt0 * 16 + n15, col1 = nt1 * 16 + n15;
            const float bv0 = bd1[col0], bv1 = bd1[col1];
#pragma unroll
            for (int r = 0; r < 8; ++r) {
                const int row = r + 8 * half;
                H1t[(col0 >> 5) * 512 + a_pack_idx(row, col0 & 31)] = f2bf(silu_f(acc0[r] + bv0));
                H1t[(col1 >> 5) * 512 + a_pack_idx(row, col1 & 31)] = f2bf(silu_f(acc1[r] + bv1));
            }
        }
        __syncthreads();
        // GEMM2: (16x256)@(256x128) -> dire (silu); mask *= dire (in place)
        {
            const int nt = wid;
            f32x8 acc = {};
            v16bf a  = load_frag_packed(H1t, lane);
            v16bf bb = load_frag_packed(Wd2s + (nt * 8) * 512, lane);
#pragma unroll
            for (int kt = 0; kt < 8; ++kt) {
                v16bf an, bn;
                if (kt < 7) {
                    an = load_frag_packed(H1t + (kt + 1) * 512, lane);
                    bn = load_frag_packed(Wd2s + (nt * 8 + kt + 1) * 512, lane);
                }
                acc = wmma_bf16(a, bb, acc);
                if (kt < 7) { a = an; bb = bn; }
            }
            const int col = nt * 16 + n15;
            const float bv = bd2[col];
#pragma unroll
            for (int r = 0; r < 8; ++r) {
                const int row = r + 8 * half, j = jt * 16 + row;
                const size_t idx = (rowB + j) * HH + col;
                maskW[idx] = f2bf(bf2f(maskW[idx]) * silu_f(acc[r] + bv));
            }
        }
        __syncthreads();
    }
}

// ---- interaction: s[bi,h] += sum_j maskD[bi,j,h]*sl[b,j,h] -----------------
__global__ void k_interact(const unsigned short* __restrict__ maskW, const float* __restrict__ slW,
                           float* __restrict__ sW) {
    const int bi = blockIdx.x, b = bi / NN;
    const int tid = threadIdx.x, hp = tid & 63, qd = tid >> 6;
    const int h0 = hp * 2;
    const size_t rowB = (size_t)bi * NN;
    float a0 = 0.f, a1 = 0.f;
    for (int j = qd * 32; j < qd * 32 + 32; ++j) {
        const unsigned int mm = *(const unsigned int*)(maskW + (rowB + j) * HH + h0);
        const float* sp = slW + ((size_t)b * NN + j) * HH + h0;
        a0 += bf2f((unsigned short)mm) * sp[0];
        a1 += bf2f((unsigned short)(mm >> 16)) * sp[1];
    }
    __shared__ float red[4][HH];
    red[qd][h0] = a0; red[qd][h0 + 1] = a1;
    __syncthreads();
    if (qd == 0) {
        float t0 = red[0][h0] + red[1][h0] + red[2][h0] + red[3][h0];
        float t1 = red[0][h0 + 1] + red[1][h0 + 1] + red[2][h0 + 1] + red[3][h0 + 1];
        sW[(size_t)bi * HH + h0]     += t0;
        sW[(size_t)bi * HH + h0 + 1] += t1;
    }
}

// ---- head: y[b] = sum_i (a[i] . Wo2 + bo2) * (z>0) -------------------------
__global__ void k_head(const float* __restrict__ aW, const float* __restrict__ Wo2,
                       const float* __restrict__ bo2, const int* __restrict__ z,
                       float* __restrict__ out) {
    const int b = blockIdx.x, t = threadIdx.x;
    const f32x4* arow = (const f32x4*)(aW + ((size_t)b * NN + t) * HH);
    const f32x4* wv = (const f32x4*)Wo2;
    float acc = 0.f;
    for (int i = 0; i < HH / 4; ++i) {
        f32x4 a = arow[i], w = wv[i];
        acc += a.x * w.x + a.y * w.y + a.z * w.z + a.w * w.w;
    }
    acc += bo2[0];
    if (z[b * NN + t] <= 0) acc = 0.f;
    __shared__ float red[NN];
    red[t] = acc; __syncthreads();
    for (int s = 64; s > 0; s >>= 1) { if (t < s) red[t] += red[t + s]; __syncthreads(); }
    if (t == 0) out[b] = red[0];
}

extern "C" void kernel_launch(void* const* d_in, const int* in_sizes, int n_in,
                              void* d_out, int out_size, void* d_ws, size_t ws_size,
                              hipStream_t stream) {
    const int*   z    = (const int*)  d_in[0];
    const float* s_in = (const float*)d_in[1];
    const float* ea   = (const float*)d_in[2];
    const float* ev   = (const float*)d_in[3];
    const float* ef   = (const float*)d_in[4];
    const float* emb  = (const float*)d_in[5];
    const float* Wef  = (const float*)d_in[6];
    const float* bef  = (const float*)d_in[7];
    const float* Ws2v = (const float*)d_in[8];
    const float* bs2v = (const float*)d_in[9];
    const float* Wq   = (const float*)d_in[10];
    const float* Wk   = (const float*)d_in[11];
    const float* Wint = (const float*)d_in[12];
    const float* bint = (const float*)d_in[13];
    const float* Wd1  = (const float*)d_in[14];
    const float* bd1  = (const float*)d_in[15];
    const float* Wd2  = (const float*)d_in[16];
    const float* bd2  = (const float*)d_in[17];
    const float* Wo1  = (const float*)d_in[18];
    const float* bo1  = (const float*)d_in[19];
    const float* Wo2  = (const float*)d_in[20];
    const float* bo2  = (const float*)d_in[21];
    float* out = (float*)d_out;

    char* ws = (char*)d_ws;
    size_t off = 0;
    auto alloc = [&](size_t bytes) -> char* {
        char* p = ws + off; off = (off + bytes + 255) & ~(size_t)255; return p;
    };
    unsigned short* wefB  = (unsigned short*)alloc((size_t)EFD * HH * 2);
    unsigned short* ws2vB = (unsigned short*)alloc((size_t)HH * HH * 2);
    unsigned short* wqB   = (unsigned short*)alloc((size_t)HH * HH * 2);
    unsigned short* wkB   = (unsigned short*)alloc((size_t)HH * HH * 2);
    unsigned short* wintB = (unsigned short*)alloc((size_t)NLAYER * HH * HH * 2);
    unsigned short* wd1B  = (unsigned short*)alloc((size_t)256 * 256 * 2);
    unsigned short* wd2B  = (unsigned short*)alloc((size_t)256 * 128 * 2);
    unsigned short* wo1B  = (unsigned short*)alloc((size_t)HH * HH * 2);
    float* snW = (float*)alloc((size_t)BB * NN * HH * 4);
    float* sW  = (float*)alloc((size_t)BB * NN * HH * 4);
    float* s1W = (float*)alloc((size_t)BB * NN * HH * 4);
    float* vW  = (float*)alloc((size_t)BB * NN * 3 * HH * 4);
    float* qW  = (float*)alloc((size_t)BB * NN * 3 * HH * 4);
    float* kW  = (float*)alloc((size_t)BB * NN * 3 * HH * 4);
    float* aW  = (float*)alloc((size_t)BB * NN * HH * 4);
    unsigned short* maskW = (unsigned short*)alloc((size_t)BB * NN * NN * HH * 2);

    auto cvt = [&](const float* src, unsigned short* dst, int K, int N) {
        k_cvt_pack<<<(K * N + 255) / 256, 256, 0, stream>>>(src, dst, K, N);
    };
    cvt(Wef,  wefB,  EFD, HH);
    cvt(Ws2v, ws2vB, HH, HH);
    cvt(Wq,   wqB,   HH, HH);
    cvt(Wk,   wkB,   HH, HH);
    for (int l = 0; l < NLAYER; ++l)
        cvt(Wint + (size_t)l * HH * HH, wintB + (size_t)l * HH * HH, HH, HH);
    cvt(Wd1,  wd1B,  256, 256);
    cvt(Wd2,  wd2B,  256, 128);
    cvt(Wo1,  wo1B,  HH, HH);

    k_sn<<<BB * NN, 128, 0, stream>>>(z, emb, snW);
    k_mask_nbr<<<BB * NN, 256, 0, stream>>>(ef, ea, s_in, snW, wefB, bef, maskW, sW);
    k_node_gemm<<<BB * NN / 16, 256, 0, stream>>>(sW, ws2vB, bs2v, s1W, 1);
    k_v<<<BB * NN, 256, 0, stream>>>(maskW, s1W, ev, vW);
    k_node_gemm<<<BB * NN * 3 / 16, 256, 0, stream>>>(vW, wqB, (const float*)nullptr, qW, 0);
    k_node_gemm<<<BB * NN * 3 / 16, 256, 0, stream>>>(vW, wkB, (const float*)nullptr, kW, 0);

    const size_t edge_lds = (size_t)(256 * 256 + 256 * 128 + 8 * 512 + 8 * 512) * 2 + 768 * 4;
    k_edge<<<BB * NN, 256, edge_lds, stream>>>(ea, ev, qW, kW, vW, wd1B, bd1, wd2B, bd2, maskW);

    for (int l = 0; l < NLAYER; ++l) {
        k_node_gemm<<<BB * NN / 16, 256, 0, stream>>>(sW, wintB + (size_t)l * HH * HH,
                                                      bint + (size_t)l * HH, s1W, 1);
        k_interact<<<BB * NN, 256, 0, stream>>>(maskW, s1W, sW);
    }

    k_node_gemm<<<BB * NN / 16, 256, 0, stream>>>(sW, wo1B, bo1, aW, 2);
    k_head<<<BB, 128, 0, stream>>>(aW, Wo2, bo2, z, out);
}
